// GlobalPointer_58523224376053
// MI455X (gfx1250) — compile-verified, matrix operand order checked by reading, not run on hardware
//
#include <hip/hip_runtime.h>

#define B_   8
#define N_   1024
#define D_   1024
#define H_   16
#define HD_  64
#define COLS (2 * H_ * HD_)   // 2048
#define NK_  32               // K-steps of 32 in kernel A
#define INF_ 1000000000000.0f

typedef __attribute__((ext_vector_type(16))) __bf16 v16bf;
typedef __attribute__((ext_vector_type(8)))  float  v8f;

union FragBF { int4 i2[2]; v16bf v; };

#if defined(__HIP_DEVICE_COMPILE__) && \
    __has_builtin(__builtin_amdgcn_global_load_async_to_lds_b128) && \
    __has_builtin(__builtin_amdgcn_s_wait_asynccnt)
#define HAVE_ASYNC 1
#else
#define HAVE_ASYNC 0
#endif

#if HAVE_ASYNC
typedef int nvec_int4 __attribute__((__vector_size__(16)));            // native int4
typedef __attribute__((address_space(1))) nvec_int4 as1_int4;          // global
typedef __attribute__((address_space(3))) nvec_int4 as3_int4;          // LDS
#endif

__device__ __forceinline__ unsigned short f2bf(float f) {
  unsigned int u = __float_as_uint(f);
  u += 0x7FFFu + ((u >> 16) & 1u);          // round-to-nearest-even
  return (unsigned short)(u >> 16);
}

// 16-byte global -> LDS copy: async-to-LDS on CDNA5, plain b128 otherwise.
__device__ __forceinline__ void cp16(unsigned short* dst_lds,
                                     const unsigned short* src_g) {
#if HAVE_ASYNC
  __builtin_amdgcn_global_load_async_to_lds_b128(
      (as1_int4*)src_g, (as3_int4*)dst_lds, 0, 0);
#else
  *(int4*)dst_lds = *(const int4*)src_g;
#endif
}

__device__ __forceinline__ void async_wait_all() {
#if HAVE_ASYNC
  __builtin_amdgcn_s_wait_asynccnt(0);
#endif
}

// ---------------------------------------------------------------------------
// Kernel 0a: x (fp32) -> xbf (bf16), row-major [8192][1024].
// ---------------------------------------------------------------------------
__global__ __launch_bounds__(256) void gp_cvt_x_kernel(
    const float* __restrict__ x, unsigned short* __restrict__ xbf) {
  size_t i = ((size_t)blockIdx.x * 256 + threadIdx.x) * 4;
  float4 v = *(const float4*)(x + i);
  uint2 o;
  o.x = (unsigned)f2bf(v.x) | ((unsigned)f2bf(v.y) << 16);
  o.y = (unsigned)f2bf(v.z) | ((unsigned)f2bf(v.w) << 16);
  *(uint2*)(xbf + i) = o;
}

// ---------------------------------------------------------------------------
// Kernel 0b: W (fp32 [1024][2048]) -> wt (bf16, transposed [2048][1024]).
// Pre-transposing makes the GEMM's B-tile a contiguous-row async copy.
// ---------------------------------------------------------------------------
__global__ __launch_bounds__(256) void gp_cvt_wt_kernel(
    const float* __restrict__ W, unsigned short* __restrict__ wt) {
  int idx = blockIdx.x * 256 + threadIdx.x;   // 0 .. 2M-1
  int k = idx >> 11;                          // / 2048
  int c = idx & 2047;
  wt[(size_t)c * D_ + k] = f2bf(W[idx]);
}

// ---------------------------------------------------------------------------
// Kernel A: proj = x @ W + b via bf16 WMMA (fp32 acc), double-buffered
// async-to-LDS staging, then interleaved RoPE, pack q/k bf16 to workspace
// as [b][h][pos][dim]. Grid: (16 col-tiles, 128 row-tiles), block 256.
// ---------------------------------------------------------------------------
__global__ __launch_bounds__(256) void gp_proj_rope_kernel(
    const unsigned short* __restrict__ xbf, const unsigned short* __restrict__ wt,
    const float* __restrict__ bias,
    unsigned short* __restrict__ qws, unsigned short* __restrict__ kws)
{
  __shared__ __align__(16) unsigned short As[2][64 * 40];    // 64 x 32K (pad 40)
  __shared__ __align__(16) unsigned short Bs[2][128 * 40];   // 128 x 32K (pad 40)

  const int tid   = threadIdx.x;
  const int lane  = tid & 31;
  const int wv    = tid >> 5;
  const int m_off = (wv & 3) << 4;
  const int n_off = (wv >> 2) << 6;
  const int frow  = lane & 15;
  const int kb    = (lane >> 4) << 3;   // ISA 16-bit fragment K split
  const int hi8   = (lane >> 4) << 3;   // +8 row offset for lanes 16..31

  const int col0 = blockIdx.x * 128;
  const int row0 = blockIdx.y * 64;

  // staging coordinates (16B chunks)
  const int ar  = tid >> 2;             // A: row 0..63
  const int ach = tid & 3;              // A: chunk 0..3 (8 bf16 each)

  v8f acc[4] = {};

  // stage tile kt into buffer bb (1 A-chunk + 2 B-chunks per thread)
  auto stage = [&](int kt, int bb) {
    const int k0 = kt * 32;
    cp16(&As[bb][ar * 40 + ach * 8],
         &xbf[(size_t)(row0 + ar) * D_ + k0 + ach * 8]);
    #pragma unroll
    for (int i = 0; i < 2; ++i) {
      int f  = tid + 256 * i;           // 0..511
      int c  = f >> 2;                  // col 0..127
      int ch = f & 3;
      cp16(&Bs[bb][c * 40 + ch * 8],
           &wt[(size_t)(col0 + c) * D_ + k0 + ch * 8]);
    }
  };

  stage(0, 0);
  async_wait_all();
  __syncthreads();

  for (int kt = 0; kt < NK_; ++kt) {
    const int cur = kt & 1;
    if (kt + 1 < NK_) stage(kt + 1, (kt + 1) & 1);   // overlap with compute

    FragBF a;
    a.i2[0] = *(const int4*)&As[cur][(m_off + frow) * 40 + kb];
    a.i2[1] = *(const int4*)&As[cur][(m_off + frow) * 40 + 16 + kb];
    #pragma unroll
    for (int j = 0; j < 4; ++j) {
      FragBF bf;
      bf.i2[0] = *(const int4*)&Bs[cur][(n_off + j * 16 + frow) * 40 + kb];
      bf.i2[1] = *(const int4*)&Bs[cur][(n_off + j * 16 + frow) * 40 + 16 + kb];
      acc[j] = __builtin_amdgcn_wmma_f32_16x16x32_bf16(
          false, a.v, false, bf.v, (short)0, acc[j], false, false);
    }

    async_wait_all();      // next tile's writes landed (this wave)
    __syncthreads();       // all waves: writes visible, reads of cur done
  }

  // --- epilogue: bias + interleaved RoPE + pack bf16 ---
  const int batch   = row0 / N_;
  const int posBase = row0 % N_;

  #pragma unroll
  for (int j = 0; j < 4; ++j) {
    int c      = col0 + n_off + j * 16 + frow;   // global column 0..2047
    float bv   = bias[c];
    int dim128 = c & 127;
    int h      = (c >> 7) & 15;
    bool isq   = dim128 < HD_;
    int  dd    = isq ? dim128 : dim128 - HD_;
    float freq = __powf(10000.0f, -(float)(dd & ~1) / (float)HD_);
    float sgn  = (dd & 1) ? 1.0f : -1.0f;
    unsigned short* dstbase =
        (isq ? qws : kws) + (size_t)(batch * H_ + h) * N_ * HD_ + dd;
    #pragma unroll
    for (int v = 0; v < 8; ++v) {
      float val     = acc[j][v] + bv;
      float partner = __shfl_xor(val, 1, 32);
      int   pos     = posBase + m_off + v + hi8;
      float s, co;
      __sincosf((float)pos * freq, &s, &co);
      float outv = val * co + sgn * partner * s;
      dstbase[(size_t)pos * HD_] = f2bf(outv);
    }
  }
}

// ---------------------------------------------------------------------------
// Kernel B: per (b,h) logits tile = q(64x64) x k^T(64x128), mask + scale.
// Async-to-LDS staging. Grid: (8 n-tiles, 16 m-tiles, 128 b*h), block 256.
// ---------------------------------------------------------------------------
__global__ __launch_bounds__(256) void gp_logits_kernel(
    const unsigned short* __restrict__ qws, const unsigned short* __restrict__ kws,
    const float* __restrict__ seq_mask, float* __restrict__ out)
{
  __shared__ __align__(16) unsigned short Qs[64 * 72];    // pad 72 (144B rows)
  __shared__ __align__(16) unsigned short Ks[128 * 72];

  const int tid   = threadIdx.x;
  const int lane  = tid & 31;
  const int wv    = tid >> 5;
  const int m_off = (wv & 3) << 4;
  const int n_off = (wv >> 2) << 6;
  const int frow  = lane & 15;
  const int kb    = (lane >> 4) << 3;
  const int hi8   = (lane >> 4) << 3;

  const int bh = blockIdx.z;
  const int b  = bh >> 4;
  const int m0 = blockIdx.y * 64;
  const int n0 = blockIdx.x * 128;

  const size_t base = (size_t)bh * N_ * HD_;
  const unsigned short* gq = qws + base + (size_t)m0 * HD_;
  const unsigned short* gk = kws + base + (size_t)n0 * HD_;

  // Q tile: 64 rows x 8 chunks = 512; K tile: 128 x 8 = 1024
  #pragma unroll
  for (int i = 0; i < 2; ++i) {
    int f = tid + 256 * i;
    int r = f >> 3, c16 = f & 7;
    cp16(&Qs[r * 72 + c16 * 8], &gq[(size_t)r * HD_ + c16 * 8]);
  }
  #pragma unroll
  for (int i = 0; i < 4; ++i) {
    int f = tid + 256 * i;
    int r = f >> 3, c16 = f & 7;
    cp16(&Ks[r * 72 + c16 * 8], &gk[(size_t)r * HD_ + c16 * 8]);
  }
  async_wait_all();
  __syncthreads();

  v8f acc[4] = {};
  #pragma unroll
  for (int s = 0; s < 2; ++s) {         // K = 64 -> two k-steps of 32
    FragBF a;
    a.i2[0] = *(const int4*)&Qs[(m_off + frow) * 72 + s * 32 + kb];
    a.i2[1] = *(const int4*)&Qs[(m_off + frow) * 72 + s * 32 + 16 + kb];
    #pragma unroll
    for (int j = 0; j < 4; ++j) {
      FragBF bf;
      bf.i2[0] = *(const int4*)&Ks[(n_off + j * 16 + frow) * 72 + s * 32 + kb];
      bf.i2[1] = *(const int4*)&Ks[(n_off + j * 16 + frow) * 72 + s * 32 + 16 + kb];
      acc[j] = __builtin_amdgcn_wmma_f32_16x16x32_bf16(
          false, a.v, false, bf.v, (short)0, acc[j], false, false);
    }
  }

  // epilogue: seq masks + strict-lower-triangle mask + 1/sqrt(64)
  float mrow[8];
  #pragma unroll
  for (int v = 0; v < 8; ++v) {
    int m = m0 + m_off + v + hi8;
    mrow[v] = seq_mask[b * N_ + m];
  }
  float* obase = out + (size_t)bh * N_ * N_;
  #pragma unroll
  for (int j = 0; j < 4; ++j) {
    int   n    = n0 + n_off + j * 16 + frow;
    float mcol = seq_mask[b * N_ + n];
    #pragma unroll
    for (int v = 0; v < 8; ++v) {
      int m = m0 + m_off + v + hi8;
      float pen = (1.0f - mrow[v]) * INF_ + (1.0f - mcol) * INF_
                + ((n < m) ? INF_ : 0.0f);
      obase[(size_t)m * N_ + n] = (acc[j][v] - pen) * 0.125f;
    }
  }
}

extern "C" void kernel_launch(void* const* d_in, const int* in_sizes, int n_in,
                              void* d_out, int out_size, void* d_ws, size_t ws_size,
                              hipStream_t stream) {
  (void)in_sizes; (void)n_in; (void)out_size; (void)ws_size;
  const float* x        = (const float*)d_in[0];
  const float* W        = (const float*)d_in[1];
  const float* bias     = (const float*)d_in[2];
  const float* seq_mask = (const float*)d_in[3];

  const size_t QK = (size_t)B_ * H_ * N_ * HD_;   // 8,388,608 elems
  unsigned short* qws = (unsigned short*)d_ws;
  unsigned short* kws = qws + QK;
  unsigned short* xbf = kws + QK;                  // 8,388,608 elems
  unsigned short* wt  = xbf + (size_t)B_ * N_ * D_; // 2,097,152 elems
  float* out = (float*)d_out;

  gp_cvt_x_kernel <<<dim3(8192), 256, 0, stream>>>(x, xbf);
  gp_cvt_wt_kernel<<<dim3(8192), 256, 0, stream>>>(W, wt);
  gp_proj_rope_kernel<<<dim3(16, 128), 256, 0, stream>>>(xbf, wt, bias, qws, kws);
  gp_logits_kernel<<<dim3(8, 16, 128), 256, 0, stream>>>(qws, kws, seq_mask, out);
}
